// Correlation_979252544003
// MI455X (gfx1250) — matrix operand order repaired and checked by reading, not compiled, not run on hardware
//
#include <hip/hip_runtime.h>
#include <hip/hip_bf16.h>

typedef __attribute__((ext_vector_type(16))) _Float16 v16h;
typedef __attribute__((ext_vector_type(8)))  _Float16 v8h;
typedef __attribute__((ext_vector_type(8)))  float    v8f;

namespace {
constexpr int kB   = 8;
constexpr int kC   = 256;
constexpr int kH   = 64;
constexpr int kW   = 96;
constexpr int kMD  = 20;     // max displacement (pixels)
constexpr int kDIL = 2;      // dilation_patch
constexpr int kPATCH = 21;   // 21x21 = 441 displacement channels
constexpr int kCPAD = kC + 8;    // 264 halves -> 528B row stride: 16B aligned, bank-skewed
constexpr int kUPAD = 144;       // W + 2*MD = 136 used, padded so the 4th u-tile never reads OOB
}

// One workgroup per (b, y). 256 threads = 8 wave32.
// LDS: x1 row as f16 [x][c] (48KB+pad), x2 row (zero-padded in u) as f16 [u][c] (76KB).
// Per dy: Gram band via v_wmma_f32_16x16x32_f16, then scatter the 21 even diagonals.
__global__ __launch_bounds__(256) void corr441_wmma(const float* __restrict__ x1,
                                                    const float* __restrict__ x2,
                                                    float* __restrict__ out) {
  __shared__ _Float16 lds1[kW * kCPAD];     // x1[b,:,y,:]  -> [x][c]
  __shared__ _Float16 lds2[kUPAD * kCPAD];  // x2[b,:,y2,:] -> [u_mem][c], u_mem = x + dx + MD

  const int bid  = blockIdx.x;          // 0 .. B*H-1
  const int b    = bid / kH;
  const int y    = bid % kH;
  const int t    = threadIdx.x;         // 0..255, doubles as the channel id during loads
  const int lane = t & 31;
  const int wave = t >> 5;              // 0..7
  const int g    = (lane >> 4) & 1;     // lane half-group (wave32)
  const int ln   = lane & 15;

  // ---------- stage x1 row once: fp32 global -> f16 LDS, transposed to [x][c] ----------
  {
    const float* src = x1 + (((size_t)b * kC + t) * kH + y) * kW;
#pragma unroll 4
    for (int x = 0; x < kW; x += 4) {
      float4 v = *(const float4*)(src + x);
      lds1[(x + 0) * kCPAD + t] = (_Float16)v.x;
      lds1[(x + 1) * kCPAD + t] = (_Float16)v.y;
      lds1[(x + 2) * kCPAD + t] = (_Float16)v.z;
      lds1[(x + 3) * kCPAD + t] = (_Float16)v.w;
    }
  }

  for (int i = 0; i < kPATCH; ++i) {         // dy = 2*i - 20
    __syncthreads();                         // lds2 WAR vs previous iteration's WMMA reads

    // ---------- stage x2 row y+dy (zero-padded in u, zero row if dy OOB) ----------
    const int y2 = y + kDIL * i - kMD;
    const bool rowok = (y2 >= 0) && (y2 < kH);
    {
      // left pad u_mem in [0, MD), right pad u_mem in [MD+W, UPAD)
#pragma unroll
      for (int um = 0; um < kMD; ++um)            lds2[um * kCPAD + t] = (_Float16)0.f;
#pragma unroll
      for (int um = kMD + kW; um < kUPAD; ++um)   lds2[um * kCPAD + t] = (_Float16)0.f;
      if (rowok) {
        const float* src = x2 + (((size_t)b * kC + t) * kH + y2) * kW;
#pragma unroll 4
        for (int x = 0; x < kW; x += 4) {
          float4 v = *(const float4*)(src + x);
          lds2[(x + kMD + 0) * kCPAD + t] = (_Float16)v.x;
          lds2[(x + kMD + 1) * kCPAD + t] = (_Float16)v.y;
          lds2[(x + kMD + 2) * kCPAD + t] = (_Float16)v.z;
          lds2[(x + kMD + 3) * kCPAD + t] = (_Float16)v.w;
        }
      } else {
        for (int x = 0; x < kW; ++x)
          lds2[(x + kMD) * kCPAD + t] = (_Float16)0.f;
      }
      // hint the next dy row toward L2 while we compute this one
      if (i + 1 < kPATCH) {
        const int y2n = y + kDIL * (i + 1) - kMD;
        if (y2n >= 0 && y2n < kH)
          __builtin_prefetch(x2 + (((size_t)b * kC + t) * kH + y2n) * kW, 0, 1);
      }
    }
    __syncthreads();                         // lds2/lds1 RAW before WMMA

    // ---------- banded Gram: 6 x-tiles x 4 u-tiles = 24 tile-GEMMs, 3 per wave ----------
#pragma unroll
    for (int r = 0; r < 3; ++r) {
      const int tg = wave * 3 + r;           // 0..23
      const int nt = tg >> 2;                // x-tile 0..5
      const int ut = tg & 3;                 // u-tile 0..3
      const int x0 = nt * 16;
      const int um0 = x0 + 16 * ut;          // u_mem tile base = (x0 - MD + 16*ut) + MD

      // A fragment (x2): A[m,k] = x2row[c0+k, u0+m]; lanes 0-15/16-31 both carry m=ln.
      // Per ISA 16-bit A layout: halves 0..7 <-> k=g*8+0..7, halves 8..15 <-> k=16+g*8+0..7.
      const _Float16* aBase = &lds2[(um0 + ln) * kCPAD];
      // B fragment (x1): B[k,n] = x1row[c0+k, x0+n]; halves 0..7 <-> k=g*8+0..7, halves 8..15 <-> k=16+g*8+0..7.
      const _Float16* bBase = &lds1[(x0 + ln) * kCPAD];

      v8f acc = {};
#pragma unroll
      for (int c0 = 0; c0 < kC; c0 += 32) {
        v8h alo = *(const v8h*)(aBase + c0 + g * 8);
        v8h ahi = *(const v8h*)(aBase + c0 + 16 + g * 8);
        v16h a = __builtin_shufflevector(alo, ahi,
                                         0, 1, 2, 3, 4, 5, 6, 7,
                                         8, 9, 10, 11, 12, 13, 14, 15);
        v8h blo = *(const v8h*)(bBase + c0 + g * 8);
        v8h bhi = *(const v8h*)(bBase + c0 + 16 + g * 8);
        v16h bm = __builtin_shufflevector(blo, bhi,
                                          0, 1, 2, 3, 4, 5, 6, 7,
                                          8, 9, 10, 11, 12, 13, 14, 15);
        acc = __builtin_amdgcn_wmma_f32_16x16x32_f16(
            /*neg_a=*/false, a, /*neg_b=*/false, bm,
            /*c_mod=*/(short)0, acc, /*reuse_a=*/false, /*reuse_b=*/false);
      }

      // ---------- scatter the even diagonals of this Gram tile ----------
      // D[m,n]: lane ln = n, VGPR rr -> m = rr + 8*g. dx = u - x = -MD + 16*ut + m - ln.
      const int x = x0 + ln;
#pragma unroll
      for (int rr = 0; rr < 8; ++rr) {
        const int m  = rr + 8 * g;
        const int dx = -kMD + 16 * ut + m - ln;
        if (((dx & 1) == 0) && dx >= -kMD && dx <= kMD) {
          const int j = (dx + kMD) >> 1;
          const int d = i * kPATCH + j;      // displacement channel, dy-major like reference
          out[(((size_t)b * (kPATCH * kPATCH) + d) * kH + y) * kW + x] = acc[rr];
        }
      }
    }
  }
}

extern "C" void kernel_launch(void* const* d_in, const int* in_sizes, int n_in,
                              void* d_out, int out_size, void* d_ws, size_t ws_size,
                              hipStream_t stream) {
  (void)in_sizes; (void)n_in; (void)d_ws; (void)ws_size; (void)out_size;
  const float* x1 = (const float*)d_in[0];
  const float* x2 = (const float*)d_in[1];
  float* out = (float*)d_out;
  dim3 grid(kB * kH);   // 512 workgroups, one per (b, y)
  dim3 block(256);      // 8 wave32
  corr441_wmma<<<grid, block, 0, stream>>>(x1, x2, out);
}
